// QDecoder_87729001988400
// MI455X (gfx1250) — matrix operand order repaired and checked by reading, not compiled, not run on hardware
//
#include <hip/hip_runtime.h>

typedef __attribute__((ext_vector_type(16))) _Float16 v16h;
typedef __attribute__((ext_vector_type(8)))  _Float16 v8h;
typedef __attribute__((ext_vector_type(8)))  float    v8f;

#define LN_EPS 1e-5f

#if __has_builtin(__builtin_amdgcn_global_load_async_to_lds_b128)
#define HAVE_ASYNC_LDS 1
typedef __attribute__((ext_vector_type(4))) int v4i_t;
typedef v4i_t __attribute__((address_space(1)))* gptr_v4i;
typedef v4i_t __attribute__((address_space(3)))* lptr_v4i;
#else
#define HAVE_ASYNC_LDS 0
#endif

__device__ __forceinline__ float wave_sum(float v) {
  #pragma unroll
  for (int off = 16; off > 0; off >>= 1) v += __shfl_down(v, off, 32);
  return v;
}

// ---------------- per-tensor abs-max (float>=0 bit-compare trick) ------------
__global__ void absmax_kernel(const float* __restrict__ W, size_t n,
                              unsigned* __restrict__ out) {
  size_t i = blockIdx.x * (size_t)blockDim.x + threadIdx.x;
  size_t stride = (size_t)gridDim.x * blockDim.x;
  float m = 0.f;
  for (; i < n; i += stride) m = fmaxf(m, fabsf(W[i]));
  #pragma unroll
  for (int off = 16; off > 0; off >>= 1) m = fmaxf(m, __shfl_down(m, off, 32));
  if ((threadIdx.x & 31) == 0) atomicMax(out, __float_as_uint(m));
}

// ------------- fake-quant int4 weights -> f16, pre-swizzled B fragments ------
// Packed layout: p = ((ct*KT + kt)*32 + lane)*16 + h
//   B element (K = kt*32 + 16*(h>>3) + 8*(lane>>4) + (h&7), N = ct*16 + (lane&15)) = Wq[o = N, k = K]
__global__ void pack_weight_kernel(const float* __restrict__ W,
                                   const unsigned* __restrict__ amax,
                                   _Float16* __restrict__ Wp, int Dout, int K) {
  float scale = __uint_as_float(*amax) * (1.0f / 7.0f);
  float inv   = 1.0f / scale;
  int KT = K >> 5;
  size_t total = (size_t)Dout * K;
  size_t i = blockIdx.x * (size_t)blockDim.x + threadIdx.x;
  size_t stride = (size_t)gridDim.x * blockDim.x;
  for (; i < total; i += stride) {
    int h    = (int)(i & 15);
    int lane = (int)((i >> 4) & 31);
    size_t t = i >> 9;
    int kt = (int)(t % KT);
    int ct = (int)(t / KT);
    int o = ct * 16 + (lane & 15);
    int k = kt * 32 + ((h >> 3) << 4) + ((lane >> 4) << 3) + (h & 7);
    float w = W[(size_t)o * K + k];
    float q = fminf(fmaxf(rintf(w * inv), -7.f), 7.f) * scale;
    Wp[i] = (_Float16)q;
  }
}

__global__ void dequant_w5_kernel(const float* __restrict__ W,
                                  const unsigned* __restrict__ amax,
                                  float* __restrict__ Wd, int n) {
  float scale = __uint_as_float(*amax) * (1.0f / 7.0f);
  float inv   = 1.0f / scale;
  int i = blockIdx.x * blockDim.x + threadIdx.x;
  if (i < n) {
    float q = fminf(fmaxf(rintf(W[i] * inv), -7.f), 7.f) * scale;
    Wd[i] = q;
  }
}

__global__ void f32_to_f16_kernel(const float* __restrict__ in,
                                  _Float16* __restrict__ out, size_t n) {
  size_t i = blockIdx.x * (size_t)blockDim.x + threadIdx.x;
  size_t stride = (size_t)gridDim.x * blockDim.x;
  for (; i < n; i += stride) out[i] = (_Float16)in[i];
}

#define WMMA_F16(Afrag, Bfrag, Cacc) \
  __builtin_amdgcn_wmma_f32_16x16x32_f16(false, (Afrag), false, (Bfrag), \
                                         (short)0, (Cacc), false, false)

// Per 16x16 output tile epilogue: bias + ReLU + uint8 fake-quant, store f16,
// and fold per-row sum/sumsq into LDS stats (shuffle across the 16-lane N
// group, then one ds_add_f32 per row per half-wave).
__device__ __forceinline__ void
epilogue_tile(v8f acc, int ct, int rt, int lane, size_t rowbase,
              const float* __restrict__ bias, float inv_s, float sv,
              _Float16* __restrict__ Out, int Dout,
              float* ldsSum, float* ldsSqs) {
  int n = ct * 16 + (lane & 15);
  float bi = bias[n];
  int mbase = rt * 16 + ((lane < 16) ? 0 : 8);
  #pragma unroll
  for (int r = 0; r < 8; ++r) {
    float v = acc[r] + bi;
    v = fmaxf(v, 0.f);
    v = fminf(rintf(v * inv_s), 255.f) * sv;   // rint == round-half-even
    Out[(rowbase + mbase + r) * (size_t)Dout + n] = (_Float16)v;
    float t1 = v, t2 = v * v;
    #pragma unroll
    for (int off = 8; off > 0; off >>= 1) {    // reduce within 16-lane group
      t1 += __shfl_xor(t1, off, 32);
      t2 += __shfl_xor(t2, off, 32);
    }
    if ((lane & 15) == 0) {
      atomicAdd(&ldsSum[mbase + r], t1);
      atomicAdd(&ldsSqs[mbase + r], t2);
    }
  }
}

// ------------- GEMM + bias + QuantReLU (quantized out) + LN stats ------------
// Workgroup = 256 thr = 8 waves owns 64 rows (4 row-tiles). Each wave runs 4
// column tiles at once over 4 row-tiles: 16 WMMAs per (4 B-frag, 4 A-frag)
// loads -> B L2 traffic 0.25 KB/WMMA and A LDS traffic 0.25 KB/WMMA (keeps a
// 64-bank LDS at/below 256 B/clk at ~1 WMMA/clk/WGP).
__global__ void __launch_bounds__(256)
qgemm_kernel(const _Float16* __restrict__ A, const _Float16* __restrict__ Wp,
             const float* __restrict__ bias, const float* __restrict__ s,
             _Float16* __restrict__ Out, float* __restrict__ stats,
             int K, int Dout) {
  extern __shared__ char smem[];
  _Float16* ldsA  = (_Float16*)smem;                     // 64*K halves, swizzled
  float*    ldsSum = (float*)(smem + (size_t)128 * K);   // 64 rows
  float*    ldsSqs = ldsSum + 64;
  const int tid  = threadIdx.x;
  const int lane = tid & 31;
  const int wave = tid >> 5;
  const int KT   = K >> 5;
  const size_t rowbase = (size_t)blockIdx.x * 64;

  if (tid < 64) { ldsSum[tid] = 0.f; ldsSqs[tid] = 0.f; }

  // Phase 1: stage 64 rows of A into the CDNA5 16-bit A-fragment swizzle.
  // Every aligned 8-half chunk of a row maps to one contiguous 16B LDS slot:
  //   kk in [0,8): lane=m,   h=0..7 ; kk in [8,16): lane=m+16, h=0..7
  //   kk in [16,24): lane=m, h=8..15; kk in [24,32): lane=m+16, h=8..15
  // Uses gfx1250 async global->LDS (ASYNCcnt) when the builtin is available.
  {
    const int chunks_per_row = K >> 3;
    for (int c = tid; c < 64 * chunks_per_row; c += 256) {
      int m  = c / chunks_per_row;
      int k  = (c - m * chunks_per_row) << 3;
      const _Float16* gsrc = A + (rowbase + m) * (size_t)K + k;
      int kt = k >> 5, kk = k & 31;
      int rt = m >> 4, mm = m & 15;
      int l  = mm + ((kk & 8) ? 16 : 0);
      int hb = (kk & 16) ? 8 : 0;
      _Float16* ldst = ldsA + ((size_t)(rt * KT + kt)) * 512 + l * 16 + hb;
#if HAVE_ASYNC_LDS
      __builtin_amdgcn_global_load_async_to_lds_b128(
          (gptr_v4i)gsrc, (lptr_v4i)ldst, 0, 0);
#else
      *(v8h*)ldst = *(const v8h*)gsrc;
#endif
    }
  }
#if HAVE_ASYNC_LDS
#if __has_builtin(__builtin_amdgcn_s_wait_asynccnt)
  __builtin_amdgcn_s_wait_asynccnt(0);
#else
  asm volatile("s_wait_asynccnt 0x0" ::: "memory");
#endif
#endif
  __syncthreads();

  const int CT = Dout >> 4;
  const int P4 = CT >> 2;            // column-tile quads
  const float sv = s[0];
  const float inv_s = 1.0f / sv;

  for (int p = wave; p < P4; p += 8) {
    const int ct0 = p, ct1 = p + P4, ct2 = p + 2 * P4, ct3 = p + 3 * P4;
    v8f c00 = {}, c01 = {}, c02 = {}, c03 = {};
    v8f c10 = {}, c11 = {}, c12 = {}, c13 = {};
    v8f c20 = {}, c21 = {}, c22 = {}, c23 = {};
    v8f c30 = {}, c31 = {}, c32 = {}, c33 = {};
    const _Float16* wp0 = Wp + (size_t)ct0 * KT * 512 + lane * 16;
    const _Float16* wp1 = Wp + (size_t)ct1 * KT * 512 + lane * 16;
    const _Float16* wp2 = Wp + (size_t)ct2 * KT * 512 + lane * 16;
    const _Float16* wp3 = Wp + (size_t)ct3 * KT * 512 + lane * 16;
    const _Float16* ap  = ldsA + lane * 16;
    #pragma unroll 2
    for (int kt = 0; kt < KT; ++kt) {
      v16h b0 = *(const v16h*)(wp0 + (size_t)kt * 512);
      v16h b1 = *(const v16h*)(wp1 + (size_t)kt * 512);
      v16h b2 = *(const v16h*)(wp2 + (size_t)kt * 512);
      v16h b3 = *(const v16h*)(wp3 + (size_t)kt * 512);
      v16h f0 = *(const v16h*)(ap + (size_t)(0 * KT + kt) * 512);
      v16h f1 = *(const v16h*)(ap + (size_t)(1 * KT + kt) * 512);
      v16h f2 = *(const v16h*)(ap + (size_t)(2 * KT + kt) * 512);
      v16h f3 = *(const v16h*)(ap + (size_t)(3 * KT + kt) * 512);
      c00 = WMMA_F16(f0, b0, c00);
      c01 = WMMA_F16(f1, b0, c01);
      c02 = WMMA_F16(f2, b0, c02);
      c03 = WMMA_F16(f3, b0, c03);
      c10 = WMMA_F16(f0, b1, c10);
      c11 = WMMA_F16(f1, b1, c11);
      c12 = WMMA_F16(f2, b1, c12);
      c13 = WMMA_F16(f3, b1, c13);
      c20 = WMMA_F16(f0, b2, c20);
      c21 = WMMA_F16(f1, b2, c21);
      c22 = WMMA_F16(f2, b2, c22);
      c23 = WMMA_F16(f3, b2, c23);
      c30 = WMMA_F16(f0, b3, c30);
      c31 = WMMA_F16(f1, b3, c31);
      c32 = WMMA_F16(f2, b3, c32);
      c33 = WMMA_F16(f3, b3, c33);
    }
    epilogue_tile(c00, ct0, 0, lane, rowbase, bias, inv_s, sv, Out, Dout, ldsSum, ldsSqs);
    epilogue_tile(c01, ct0, 1, lane, rowbase, bias, inv_s, sv, Out, Dout, ldsSum, ldsSqs);
    epilogue_tile(c02, ct0, 2, lane, rowbase, bias, inv_s, sv, Out, Dout, ldsSum, ldsSqs);
    epilogue_tile(c03, ct0, 3, lane, rowbase, bias, inv_s, sv, Out, Dout, ldsSum, ldsSqs);
    epilogue_tile(c10, ct1, 0, lane, rowbase, bias, inv_s, sv, Out, Dout, ldsSum, ldsSqs);
    epilogue_tile(c11, ct1, 1, lane, rowbase, bias, inv_s, sv, Out, Dout, ldsSum, ldsSqs);
    epilogue_tile(c12, ct1, 2, lane, rowbase, bias, inv_s, sv, Out, Dout, ldsSum, ldsSqs);
    epilogue_tile(c13, ct1, 3, lane, rowbase, bias, inv_s, sv, Out, Dout, ldsSum, ldsSqs);
    epilogue_tile(c20, ct2, 0, lane, rowbase, bias, inv_s, sv, Out, Dout, ldsSum, ldsSqs);
    epilogue_tile(c21, ct2, 1, lane, rowbase, bias, inv_s, sv, Out, Dout, ldsSum, ldsSqs);
    epilogue_tile(c22, ct2, 2, lane, rowbase, bias, inv_s, sv, Out, Dout, ldsSum, ldsSqs);
    epilogue_tile(c23, ct2, 3, lane, rowbase, bias, inv_s, sv, Out, Dout, ldsSum, ldsSqs);
    epilogue_tile(c30, ct3, 0, lane, rowbase, bias, inv_s, sv, Out, Dout, ldsSum, ldsSqs);
    epilogue_tile(c31, ct3, 1, lane, rowbase, bias, inv_s, sv, Out, Dout, ldsSum, ldsSqs);
    epilogue_tile(c32, ct3, 2, lane, rowbase, bias, inv_s, sv, Out, Dout, ldsSum, ldsSqs);
    epilogue_tile(c33, ct3, 3, lane, rowbase, bias, inv_s, sv, Out, Dout, ldsSum, ldsSqs);
  }
  __syncthreads();

  // Finalize per-row LN stats (mu, 1/sigma).
  if (tid < 64) {
    float mu  = ldsSum[tid] / (float)Dout;
    float var = ldsSqs[tid] / (float)Dout - mu * mu;
    float inv = rsqrtf(var + LN_EPS);
    size_t row = rowbase + tid;
    stats[row * 2 + 0] = mu;
    stats[row * 2 + 1] = inv;
  }
}

// ---------------- in-place LayerNorm apply (one block per row) ---------------
__global__ void __launch_bounds__(256)
ln_apply_kernel(_Float16* __restrict__ buf, const float* __restrict__ stats,
                const float* __restrict__ g, const float* __restrict__ be,
                int Dout) {
  size_t row = blockIdx.x;
  float mu  = stats[row * 2 + 0];
  float inv = stats[row * 2 + 1];
  _Float16* p = buf + row * (size_t)Dout;
  for (int n = threadIdx.x; n < Dout; n += 256) {
    float v = (float)p[n];
    p[n] = (_Float16)((v - mu) * inv * g[n] + be[n]);
  }
}

// ---------------- final matvec: [N,1024] @ w5 + b5 -> [N,1] ------------------
__global__ void __launch_bounds__(256)
final_kernel(const _Float16* __restrict__ A, const float* __restrict__ Wd,
             const float* __restrict__ b5, float* __restrict__ out, int K) {
  int lane = threadIdx.x & 31;
  int wave = threadIdx.x >> 5;
  size_t row = (size_t)blockIdx.x * 8 + wave;
  const _Float16* ap = A + row * (size_t)K;
  float sum = 0.f;
  for (int k = lane; k < K; k += 32) sum += (float)ap[k] * Wd[k];
  sum = wave_sum(sum);
  if (lane == 0) out[row] = sum + b5[0];
}

extern "C" void kernel_launch(void* const* d_in, const int* in_sizes, int n_in,
                              void* d_out, int out_size, void* d_ws, size_t ws_size,
                              hipStream_t stream) {
  (void)in_sizes; (void)n_in; (void)out_size; (void)ws_size;
  const float* x  = (const float*)d_in[0];
  const float* W1 = (const float*)d_in[1];
  const float* b1 = (const float*)d_in[2];
  const float* W2 = (const float*)d_in[3];
  const float* b2 = (const float*)d_in[4];
  const float* W3 = (const float*)d_in[5];
  const float* b3 = (const float*)d_in[6];
  const float* W4 = (const float*)d_in[7];
  const float* b4 = (const float*)d_in[8];
  const float* W5 = (const float*)d_in[9];
  const float* b5 = (const float*)d_in[10];
  const float* g1 = (const float*)d_in[11];
  const float* be1= (const float*)d_in[12];
  const float* g2 = (const float*)d_in[13];
  const float* be2= (const float*)d_in[14];
  const float* g3 = (const float*)d_in[15];
  const float* be3= (const float*)d_in[16];
  const float* g4 = (const float*)d_in[17];
  const float* be4= (const float*)d_in[18];
  const float* s1 = (const float*)d_in[19];
  const float* s2 = (const float*)d_in[20];
  const float* s3 = (const float*)d_in[21];
  const float* s4 = (const float*)d_in[22];

  const int N = 32768, D = 1024;

  // -------- workspace layout --------
  char* ws = (char*)d_ws;
  unsigned* amax = (unsigned*)ws;              // 5 slots
  size_t off = 256;
  const size_t wpack_sz = (size_t)2048 * 1024 * sizeof(_Float16); // 4 MB each
  _Float16* W1p = (_Float16*)(ws + off); off += wpack_sz;
  _Float16* W2p = (_Float16*)(ws + off); off += wpack_sz;
  _Float16* W3p = (_Float16*)(ws + off); off += wpack_sz;
  _Float16* W4p = (_Float16*)(ws + off); off += wpack_sz;
  float*    W5d = (float*)(ws + off);    off += 4096;
  float*    stats = (float*)(ws + off);  off += (size_t)N * 2 * sizeof(float); // 256 KB
  off = (off + 255) & ~(size_t)255;
  _Float16* bufA = (_Float16*)(ws + off); off += (size_t)N * 1024 * sizeof(_Float16); // 64 MB
  _Float16* bufB = (_Float16*)(ws + off);                                             // 128 MB

  // -------- weight quantization prep --------
  (void)hipMemsetAsync(amax, 0, 8 * sizeof(unsigned), stream);
  absmax_kernel<<<256, 256, 0, stream>>>(W1, (size_t)2048 * 1024, amax + 0);
  absmax_kernel<<<256, 256, 0, stream>>>(W2, (size_t)1024 * 2048, amax + 1);
  absmax_kernel<<<256, 256, 0, stream>>>(W3, (size_t)2048 * 1024, amax + 2);
  absmax_kernel<<<256, 256, 0, stream>>>(W4, (size_t)1024 * 2048, amax + 3);
  absmax_kernel<<<8,   256, 0, stream>>>(W5, (size_t)1024,        amax + 4);

  pack_weight_kernel<<<2048, 256, 0, stream>>>(W1, amax + 0, W1p, 2048, 1024);
  pack_weight_kernel<<<2048, 256, 0, stream>>>(W2, amax + 1, W2p, 1024, 2048);
  pack_weight_kernel<<<2048, 256, 0, stream>>>(W3, amax + 2, W3p, 2048, 1024);
  pack_weight_kernel<<<2048, 256, 0, stream>>>(W4, amax + 3, W4p, 1024, 2048);
  dequant_w5_kernel<<<4, 256, 0, stream>>>(W5, amax + 4, W5d, 1024);

  f32_to_f16_kernel<<<4096, 256, 0, stream>>>(x, bufA, (size_t)N * D);

  // -------- fused blocks: qgemm (+stats) then in-place LN apply --------
  (void)hipFuncSetAttribute((const void*)qgemm_kernel,
                            hipFuncAttributeMaxDynamicSharedMemorySize, 288 * 1024);

  const size_t sm_k1024 = (size_t)128 * 1024 + 512; // 64 rows * 1024 halves * 2B + stats
  const size_t sm_k2048 = (size_t)256 * 1024 + 512;

  qgemm_kernel<<<N / 64, 256, sm_k1024, stream>>>(bufA, W1p, b1, s1, bufB, stats, 1024, 2048);
  ln_apply_kernel<<<N, 256, 0, stream>>>(bufB, stats, g1, be1, 2048);

  qgemm_kernel<<<N / 64, 256, sm_k2048, stream>>>(bufB, W2p, b2, s2, bufA, stats, 2048, 1024);
  ln_apply_kernel<<<N, 256, 0, stream>>>(bufA, stats, g2, be2, 1024);

  qgemm_kernel<<<N / 64, 256, sm_k1024, stream>>>(bufA, W3p, b3, s3, bufB, stats, 1024, 2048);
  ln_apply_kernel<<<N, 256, 0, stream>>>(bufB, stats, g3, be3, 2048);

  qgemm_kernel<<<N / 64, 256, sm_k2048, stream>>>(bufB, W4p, b4, s4, bufA, stats, 2048, 1024);
  ln_apply_kernel<<<N, 256, 0, stream>>>(bufA, stats, g4, be4, 1024);

  final_kernel<<<N / 8, 256, 0, stream>>>(bufA, W5d, b5, (float*)d_out, 1024);
}
